// IEGMN_16234976379301
// MI455X (gfx1250) — matrix operand order repaired and checked by reading, not compiled, not run on hardware
//
#include <hip/hip_runtime.h>
#include <hip/hip_bf16.h>
#include <stdint.h>

#define N_LIG 4096
#define N_REC 8192
#define E_LIG 65536
#define E_REC 262144
#define F_LIG 15
#define F_REC 27

typedef __attribute__((ext_vector_type(16))) __bf16 v16bf;
typedef __attribute__((ext_vector_type(8)))  float  v8f;

__constant__ float c_sig[15] = {
  1.f, 1.5f, 2.25f, 3.375f, 5.0625f, 7.59375f, 11.390625f, 17.0859375f,
  25.62890625f, 38.443359375f, 57.6650390625f, 86.49755859375f,
  129.746337890625f, 194.6195068359375f, 291.92926025390625f };

__device__ __forceinline__ float lrelu(float x) { return x > 0.f ? x : 0.01f * x; }

__device__ __forceinline__ __bf16 f2bf(float x) {
  union { float f; unsigned u; } v; v.f = x;
  unsigned r = v.u + 0x7FFFu + ((v.u >> 16) & 1u);
  unsigned short h = (unsigned short)(r >> 16);
  __bf16 o; __builtin_memcpy(&o, &h, 2); return o;
}

__device__ __forceinline__ v8f wmma_bf16(v16bf a, v16bf b, v8f c) {
  return __builtin_amdgcn_wmma_f32_16x16x32_bf16(false, a, false, b, (short)0, c,
                                                 false, false);
}

// A-fragment (16x32 bf16, M=lane%16): element e -> K = 32*kt + 16*half + e
__device__ __forceinline__ v16bf load_a_frag(const __bf16* row, int kt, int half) {
  const __bf16* p = row + kt * 32 + half * 16;
  v16bf o;
#pragma unroll
  for (int i = 0; i < 8; ++i) o[i] = p[i];
#pragma unroll
  for (int i = 0; i < 8; ++i) o[8 + i] = p[8 + i];
  return o;
}

// B-fragment from 16 contiguous bf16 (K-contiguous for column n=lane%16)
__device__ __forceinline__ v16bf load16(const __bf16* p) {
  v16bf o;
#pragma unroll
  for (int i = 0; i < 16; ++i) o[i] = p[i];
  return o;
}

__device__ __forceinline__ float red_sum16(float v) {
#pragma unroll
  for (int off = 1; off < 16; off <<= 1) v += __shfl_xor(v, off, 32);
  return v;
}
__device__ __forceinline__ float red_max16(float v) {
#pragma unroll
  for (int off = 1; off < 16; off <<= 1) v = fmaxf(v, __shfl_xor(v, off, 32));
  return v;
}

__device__ __forceinline__ void zero4(v8f a[4]) {
#pragma unroll
  for (int nt = 0; nt < 4; ++nt)
#pragma unroll
    for (int r = 0; r < 8; ++r) a[nt][r] = 0.f;
}

// bias + leakyrelu + layernorm over 64 features in WMMA C-layout:
// lane holds (row = r + 8*half, feature = nt*16 + lane%16)
__device__ __forceinline__ void bias_lrelu_ln(v8f acc[4], const float* b, const float* g,
                                              const float* bt, int lane, float out[4][8]) {
  const int m = lane & 15;
  float vals[4][8], gg[4], bb[4];
#pragma unroll
  for (int nt = 0; nt < 4; ++nt) {
    float bias = b[nt * 16 + m];
    gg[nt] = g[nt * 16 + m]; bb[nt] = bt[nt * 16 + m];
#pragma unroll
    for (int r = 0; r < 8; ++r) vals[nt][r] = lrelu(acc[nt][r] + bias);
  }
#pragma unroll
  for (int r = 0; r < 8; ++r) {
    float s = vals[0][r] + vals[1][r] + vals[2][r] + vals[3][r];
    s = red_sum16(s);
    float mean = s * 0.015625f;
    float q = 0.f;
#pragma unroll
    for (int nt = 0; nt < 4; ++nt) { float d = vals[nt][r] - mean; q += d * d; }
    q = red_sum16(q);
    float inv = rsqrtf(q * 0.015625f + 1e-5f);
#pragma unroll
    for (int nt = 0; nt < 4; ++nt) out[nt][r] = (vals[nt][r] - mean) * inv * gg[nt] + bb[nt];
  }
}

// ---------------- utility kernels ----------------

__global__ void zero_kernel(float* p, int n) {
  int i = blockIdx.x * blockDim.x + threadIdx.x;
  if (i < n) p[i] = 0.f;
}

__global__ void f2bf_kernel(const float* __restrict__ in, __bf16* __restrict__ out, int n) {
  int i = blockIdx.x * blockDim.x + threadIdx.x;
  if (i < n) out[i] = f2bf(in[i]);
}

// Pack W[K,64] (f32 row-major) into wave32 B-fragment layout:
// frag[(kt*4+nt)*32 + lane][e] = bf16(W[32*kt + 16*(lane/16) + e][16*nt + lane%16])
__global__ void pack_wfrag(const float* __restrict__ W, int Korig, int ktiles,
                           __bf16* __restrict__ out) {
  int tid = blockIdx.x * blockDim.x + threadIdx.x;
  int total = ktiles * 2048;
  if (tid >= total) return;
  int e = tid & 15, lane = (tid >> 4) & 31, nt = (tid >> 9) & 3, kt = tid >> 11;
  int K = 32 * kt + 16 * (lane >> 4) + e;
  int n = (lane & 15) + 16 * nt;
  float v = (K < Korig) ? W[K * 64 + n] : 0.f;
  out[tid] = f2bf(v);
}

// ---------------- QKV projection (one wave = 16 nodes) ----------------

__global__ void qkv_kernel(const __bf16* __restrict__ hbf, int N,
                           const v16bf* __restrict__ qw, const v16bf* __restrict__ kw,
                           const v16bf* __restrict__ vw,
                           __bf16* __restrict__ qo, __bf16* __restrict__ ko,
                           __bf16* __restrict__ vT) {
  __shared__ __bf16 sh[16 * 64];
  const int lane = threadIdx.x & 31, half = lane >> 4, m = lane & 15;
  const int nb = blockIdx.x * 16;
  for (int idx = lane; idx < 16 * 64; idx += 32)
    sh[idx] = hbf[(nb + (idx >> 6)) * 64 + (idx & 63)];
  __syncthreads();
  const v16bf a0 = load_a_frag(sh + m * 64, 0, half);
  const v16bf a1 = load_a_frag(sh + m * 64, 1, half);
  v8f aq[4], ak[4], av[4];
  zero4(aq); zero4(ak); zero4(av);
#pragma unroll
  for (int nt = 0; nt < 4; ++nt) {
    aq[nt] = wmma_bf16(a0, qw[nt * 32 + lane], aq[nt]);
    aq[nt] = wmma_bf16(a1, qw[(4 + nt) * 32 + lane], aq[nt]);
    ak[nt] = wmma_bf16(a0, kw[nt * 32 + lane], ak[nt]);
    ak[nt] = wmma_bf16(a1, kw[(4 + nt) * 32 + lane], ak[nt]);
    av[nt] = wmma_bf16(a0, vw[nt * 32 + lane], av[nt]);
    av[nt] = wmma_bf16(a1, vw[(4 + nt) * 32 + lane], av[nt]);
  }
#pragma unroll
  for (int nt = 0; nt < 4; ++nt)
#pragma unroll
    for (int r = 0; r < 8; ++r) {
      int row = nb + r + 8 * half, f = nt * 16 + m;
      qo[row * 64 + f] = f2bf(lrelu(aq[nt][r]));
      ko[row * 64 + f] = f2bf(lrelu(ak[nt][r]));
      vT[(long long)f * N + row] = f2bf(av[nt][r]);
    }
}

// ---------------- fused masked flash attention (one wave = 16 query rows) ----------------
// Mask tile is staged through LDS with memory-order-coalesced loads; the next chunk's
// mask region is software-prefetched (global_prefetch_b8) while WMMAs run.

__global__ void attn_kernel(const __bf16* __restrict__ q, const __bf16* __restrict__ kmat,
                            const __bf16* __restrict__ vT, int Nk,
                            const float* __restrict__ mask, long long srow, long long scol,
                            float* __restrict__ out) {
  __shared__ __bf16 shq[16 * 64];
  __shared__ __bf16 shp[16 * 32];
  __shared__ float shm[16 * 32];
  const int lane = threadIdx.x & 31, half = lane >> 4, m = lane & 15;
  const int rowbase = blockIdx.x * 16;
  for (int idx = lane; idx < 16 * 64; idx += 32)
    shq[idx] = q[(rowbase + (idx >> 6)) * 64 + (idx & 63)];
  __syncthreads();
  const v16bf qa0 = load_a_frag(shq + m * 64, 0, half);
  const v16bf qa1 = load_a_frag(shq + m * 64, 1, half);
  v8f acc[4]; zero4(acc);
  float rmax[8], rsum[8];
#pragma unroll
  for (int r = 0; r < 8; ++r) { rmax[r] = -1e30f; rsum[r] = 0.f; }

  for (int cb = 0; cb < Nk; cb += 32) {
    // --- stage 16x32 mask tile into LDS, coalesced along the unit-stride dim ---
    if (scol == 1) {
#pragma unroll
      for (int r = 0; r < 16; ++r)
        shm[r * 32 + lane] = mask[(long long)(rowbase + r) * srow + (cb + lane)];
      if (cb + 32 < Nk)  // prefetch next tile: one line-touch per lane
        __builtin_prefetch(&mask[(long long)(rowbase + (lane >> 1)) * srow + cb + 32 +
                                 (lane & 1) * 16], 0, 3);
    } else {
#pragma unroll
      for (int j = 0; j < 16; ++j) {
        int c = j * 2 + half, r = m;
        shm[r * 32 + c] = mask[(long long)(cb + c) * scol + (rowbase + r)];
      }
      if (cb + 32 < Nk)
        __builtin_prefetch(&mask[(long long)(cb + 32 + lane) * scol + rowbase], 0, 3);
    }

    v8f s[2];
#pragma unroll
    for (int nt = 0; nt < 2; ++nt) {
#pragma unroll
      for (int r = 0; r < 8; ++r) s[nt][r] = 0.f;
      const int col = cb + nt * 16 + m;
      s[nt] = wmma_bf16(qa0, load16(kmat + col * 64 + 16 * half), s[nt]);
      s[nt] = wmma_bf16(qa1, load16(kmat + col * 64 + 32 + 16 * half), s[nt]);
    }
    float sv[2][8], tmax[8];
#pragma unroll
    for (int r = 0; r < 8; ++r) tmax[r] = rmax[r];
#pragma unroll
    for (int nt = 0; nt < 2; ++nt) {
#pragma unroll
      for (int r = 0; r < 8; ++r) {
        float mk = shm[(r + 8 * half) * 32 + nt * 16 + m];
        float x = mk * s[nt][r] - 1000.f * (1.f - mk);
        sv[nt][r] = x;
        tmax[r] = fmaxf(tmax[r], x);
      }
    }
    float scale[8];
#pragma unroll
    for (int r = 0; r < 8; ++r) {
      float mn = red_max16(tmax[r]);
      scale[r] = __expf(rmax[r] - mn);
      rmax[r] = mn;
    }
    float ps[8];
#pragma unroll
    for (int r = 0; r < 8; ++r) ps[r] = 0.f;
#pragma unroll
    for (int nt = 0; nt < 2; ++nt)
#pragma unroll
      for (int r = 0; r < 8; ++r) {
        float p = __expf(sv[nt][r] - rmax[r]);
        shp[(r + 8 * half) * 32 + nt * 16 + m] = f2bf(p);
        ps[r] += p;
      }
#pragma unroll
    for (int r = 0; r < 8; ++r) {
      rsum[r] = rsum[r] * scale[r] + red_sum16(ps[r]);
#pragma unroll
      for (int nt = 0; nt < 4; ++nt) acc[nt][r] = acc[nt][r] * scale[r];
    }
    const v16bf pa = load_a_frag(shp + m * 32, 0, half);
#pragma unroll
    for (int nt = 0; nt < 4; ++nt)
      acc[nt] = wmma_bf16(pa, load16(vT + (long long)(nt * 16 + m) * Nk + cb + 16 * half),
                          acc[nt]);
  }
#pragma unroll
  for (int nt = 0; nt < 4; ++nt)
#pragma unroll
    for (int r = 0; r < 8; ++r)
      out[(rowbase + r + 8 * half) * 64 + nt * 16 + m] = acc[nt][r] / rsum[r];
}

// ---------------- fused edge-message MLP + coords MLP + scatter (one wave = 16 edges) ------

__global__ void edge_kernel(int F, const int* __restrict__ src, const int* __restrict__ dst,
                            const float* __restrict__ efeat, const __bf16* __restrict__ hbf,
                            const float* __restrict__ coords,
                            const v16bf* __restrict__ w1f, int kt1,
                            const float* __restrict__ b1, const float* __restrict__ g,
                            const float* __restrict__ bt,
                            const v16bf* __restrict__ w2f, const float* __restrict__ b2,
                            const v16bf* __restrict__ cw1f, const float* __restrict__ cb1,
                            const float* __restrict__ cw2, const float* __restrict__ cb2,
                            float* __restrict__ aggr, float* __restrict__ xacc,
                            float* __restrict__ cnt) {
  __shared__ __bf16 shA[16 * 192];
  __shared__ __bf16 shB[16 * 64];
  __shared__ float shd2[16];
  __shared__ int shs[16], shd[16];
  const int lane = threadIdx.x & 31, half = lane >> 4, m = lane & 15;
  const int ebase = blockIdx.x * 16;
  if (lane < 16) {
    int e = ebase + lane;
    int s = src[e], d = dst[e];
    shs[lane] = s; shd[lane] = d;
    float dx = coords[s * 3 + 0] - coords[d * 3 + 0];
    float dy = coords[s * 3 + 1] - coords[d * 3 + 1];
    float dz = coords[s * 3 + 2] - coords[d * 3 + 2];
    shd2[lane] = dx * dx + dy * dy + dz * dz;
  }
  __syncthreads();
  const int FT = 128 + F + 15;
  for (int idx = lane; idx < 16 * 192; idx += 32) {
    int row = idx / 192, k = idx % 192;
    __bf16 v;
    if (k < 64)            v = hbf[shs[row] * 64 + k];
    else if (k < 128)      v = hbf[shd[row] * 64 + (k - 64)];
    else if (k < 128 + F)  v = f2bf(efeat[(long long)(ebase + row) * F + (k - 128)]);
    else if (k < FT)       v = f2bf(__expf(-shd2[row] / c_sig[k - 128 - F]));
    else                   v = f2bf(0.f);
    shA[row * 192 + k] = v;
  }
  __syncthreads();

  v8f acc[4]; zero4(acc);
  for (int kt = 0; kt < kt1; ++kt) {
    v16bf a = load_a_frag(shA + m * 192, kt, half);
#pragma unroll
    for (int nt = 0; nt < 4; ++nt) acc[nt] = wmma_bf16(a, w1f[(kt * 4 + nt) * 32 + lane], acc[nt]);
  }
  float nrm[4][8];
  bias_lrelu_ln(acc, b1, g, bt, lane, nrm);
#pragma unroll
  for (int nt = 0; nt < 4; ++nt)
#pragma unroll
    for (int r = 0; r < 8; ++r) shB[(r + 8 * half) * 64 + nt * 16 + m] = f2bf(nrm[nt][r]);

  v8f acc2[4]; zero4(acc2);
#pragma unroll
  for (int kt = 0; kt < 2; ++kt) {
    v16bf a = load_a_frag(shB + m * 64, kt, half);
#pragma unroll
    for (int nt = 0; nt < 4; ++nt) acc2[nt] = wmma_bf16(a, w2f[(kt * 4 + nt) * 32 + lane], acc2[nt]);
  }
  float msgv[4][8];
#pragma unroll
  for (int nt = 0; nt < 4; ++nt) {
    float bb = b2[nt * 16 + m];
#pragma unroll
    for (int r = 0; r < 8; ++r) msgv[nt][r] = acc2[nt][r] + bb;
  }
  // coords MLP: coef = lrelu(msg @ cw1 + cb1) @ cw2 + cb2
#pragma unroll
  for (int nt = 0; nt < 4; ++nt)
#pragma unroll
    for (int r = 0; r < 8; ++r) shB[(r + 8 * half) * 64 + nt * 16 + m] = f2bf(msgv[nt][r]);
  v8f acc3[4]; zero4(acc3);
#pragma unroll
  for (int kt = 0; kt < 2; ++kt) {
    v16bf a = load_a_frag(shB + m * 64, kt, half);
#pragma unroll
    for (int nt = 0; nt < 4; ++nt) acc3[nt] = wmma_bf16(a, cw1f[(kt * 4 + nt) * 32 + lane], acc3[nt]);
  }
  float cp[8];
#pragma unroll
  for (int r = 0; r < 8; ++r) cp[r] = 0.f;
#pragma unroll
  for (int nt = 0; nt < 4; ++nt) {
    float w = cw2[nt * 16 + m], bb = cb1[nt * 16 + m];
#pragma unroll
    for (int r = 0; r < 8; ++r) cp[r] += lrelu(acc3[nt][r] + bb) * w;
  }
  float cb2v = cb2[0];
  float coef[8];
#pragma unroll
  for (int r = 0; r < 8; ++r) coef[r] = red_sum16(cp[r]) + cb2v;

  // scatter: segment sums via atomics
#pragma unroll
  for (int r = 0; r < 8; ++r) {
    int dd = shd[r + 8 * half];
#pragma unroll
    for (int nt = 0; nt < 4; ++nt)
      atomicAdd(&aggr[(long long)dd * 64 + nt * 16 + m], msgv[nt][r]);
  }
  if (m < 3) {
#pragma unroll
    for (int r = 0; r < 8; ++r) {
      int row = r + 8 * half;
      int s = shs[row], dd = shd[row];
      float xr = coords[s * 3 + m] - coords[dd * 3 + m];
      atomicAdd(&xacc[dd * 3 + m], xr * coef[r]);
    }
  }
  if (m == 0) {
#pragma unroll
    for (int r = 0; r < 8; ++r) atomicAdd(&cnt[shd[r + 8 * half]], 1.f);
  }
}

// ---------------- node update MLP (one wave = 16 nodes) ----------------

__global__ void node_kernel(const float* __restrict__ h, const float* __restrict__ aggr,
                            const float* __restrict__ cnt, const float* __restrict__ att,
                            const float* __restrict__ origh,
                            const v16bf* __restrict__ w1f, const float* __restrict__ b1,
                            const float* __restrict__ g, const float* __restrict__ bt,
                            const v16bf* __restrict__ w2f, const float* __restrict__ b2,
                            float* __restrict__ hout) {
  __shared__ __bf16 shA[16 * 256];
  __shared__ __bf16 shB[16 * 64];
  __shared__ float shic[16];
  const int lane = threadIdx.x & 31, half = lane >> 4, m = lane & 15;
  const int nb = blockIdx.x * 16;
  if (lane < 16) shic[lane] = 1.f / fmaxf(cnt[nb + lane], 1.f);
  __syncthreads();
  for (int idx = lane; idx < 16 * 256; idx += 32) {
    int row = idx >> 8, k = idx & 255;
    long long n = nb + row;
    float v;
    if (k < 64)       v = h[n * 64 + k];
    else if (k < 128) v = aggr[n * 64 + (k - 64)] * shic[row];
    else if (k < 192) v = att[n * 64 + (k - 128)];
    else              v = origh[n * 64 + (k - 192)];
    shA[row * 256 + k] = f2bf(v);
  }
  __syncthreads();
  v8f acc[4]; zero4(acc);
#pragma unroll
  for (int kt = 0; kt < 8; ++kt) {
    v16bf a = load_a_frag(shA + m * 256, kt, half);
#pragma unroll
    for (int nt = 0; nt < 4; ++nt) acc[nt] = wmma_bf16(a, w1f[(kt * 4 + nt) * 32 + lane], acc[nt]);
  }
  float nrm[4][8];
  bias_lrelu_ln(acc, b1, g, bt, lane, nrm);
#pragma unroll
  for (int nt = 0; nt < 4; ++nt)
#pragma unroll
    for (int r = 0; r < 8; ++r) shB[(r + 8 * half) * 64 + nt * 16 + m] = f2bf(nrm[nt][r]);
  v8f acc2[4]; zero4(acc2);
#pragma unroll
  for (int kt = 0; kt < 2; ++kt) {
    v16bf a = load_a_frag(shB + m * 64, kt, half);
#pragma unroll
    for (int nt = 0; nt < 4; ++nt) acc2[nt] = wmma_bf16(a, w2f[(kt * 4 + nt) * 32 + lane], acc2[nt]);
  }
#pragma unroll
  for (int nt = 0; nt < 4; ++nt) {
    float bb = b2[nt * 16 + m];
#pragma unroll
    for (int r = 0; r < 8; ++r) {
      long long row = nb + r + 8 * half;
      float res = acc2[nt][r] + bb;
      hout[row * 64 + nt * 16 + m] = 0.5f * res + 0.5f * h[row * 64 + nt * 16 + m];
    }
  }
}

// ---------------- coordinate finalize ----------------

__global__ void coords_kernel(int N, const float* __restrict__ origc, const float* __restrict__ c,
                              const float* __restrict__ xacc, const float* __restrict__ cnt,
                              float* __restrict__ out) {
  int i = blockIdx.x * blockDim.x + threadIdx.x;
  if (i >= N * 3) return;
  float inv = 1.f / fmaxf(cnt[i / 3], 1.f);
  out[i] = 0.25f * origc[i] + 0.75f * c[i] + xacc[i] * inv;
}

// ---------------- host ----------------

extern "C" void kernel_launch(void* const* d_in, const int* in_sizes, int n_in,
                              void* d_out, int out_size, void* d_ws, size_t ws_size,
                              hipStream_t stream) {
  (void)in_sizes; (void)n_in; (void)out_size; (void)ws_size;
  const float* coords_lig = (const float*)d_in[0];
  const float* h_lig      = (const float*)d_in[1];
  const float* orig_h_lig = (const float*)d_in[2];
  const float* orig_c_lig = (const float*)d_in[3];
  const float* coords_rec = (const float*)d_in[4];
  const float* h_rec      = (const float*)d_in[5];
  const float* orig_h_rec = (const float*)d_in[6];
  const float* orig_c_rec = (const float*)d_in[7];
  const int* lig_src = (const int*)d_in[8];
  const int* lig_dst = (const int*)d_in[9];
  const float* lig_efeat = (const float*)d_in[10];
  const int* rec_src = (const int*)d_in[11];
  const int* rec_dst = (const int*)d_in[12];
  const float* rec_efeat = (const float*)d_in[13];
  const float* mask = (const float*)d_in[14];

  char* ws = (char*)d_ws;
  size_t off = 0;
  auto alloc = [&](size_t bytes) -> void* {
    void* p = ws + off;
    off = (off + bytes + 255) & ~(size_t)255;
    return p;
  };
  // accumulators first (zeroed each launch)
  float* aggr_l = (float*)alloc((size_t)N_LIG * 64 * 4);
  float* aggr_r = (float*)alloc((size_t)N_REC * 64 * 4);
  float* xacc_l = (float*)alloc((size_t)N_LIG * 3 * 4);
  float* xacc_r = (float*)alloc((size_t)N_REC * 3 * 4);
  float* cnt_l  = (float*)alloc((size_t)N_LIG * 4);
  float* cnt_r  = (float*)alloc((size_t)N_REC * 4);
  const size_t zero_floats = off / 4;

  __bf16* hbf_l = (__bf16*)alloc((size_t)N_LIG * 64 * 2);
  __bf16* hbf_r = (__bf16*)alloc((size_t)N_REC * 64 * 2);
  __bf16* q_l = (__bf16*)alloc((size_t)N_LIG * 64 * 2);
  __bf16* k_l = (__bf16*)alloc((size_t)N_LIG * 64 * 2);
  __bf16* vT_l = (__bf16*)alloc((size_t)N_LIG * 64 * 2);
  __bf16* q_r = (__bf16*)alloc((size_t)N_REC * 64 * 2);
  __bf16* k_r = (__bf16*)alloc((size_t)N_REC * 64 * 2);
  __bf16* vT_r = (__bf16*)alloc((size_t)N_REC * 64 * 2);
  float* att_l = (float*)alloc((size_t)N_LIG * 64 * 4);
  float* att_r = (float*)alloc((size_t)N_REC * 64 * 4);

  auto fragalloc = [&](int ktiles) { return alloc((size_t)ktiles * 2048 * 2); };
  void* w1f_l = fragalloc(6);  void* w1f_r = fragalloc(6);
  void* w2f_l = fragalloc(2);  void* w2f_r = fragalloc(2);
  void* cw1f_l = fragalloc(2); void* cw1f_r = fragalloc(2);
  void* qwf_l = fragalloc(2);  void* kwf_l = fragalloc(2);  void* vwf_l = fragalloc(2);
  void* qwf_r = fragalloc(2);  void* kwf_r = fragalloc(2);  void* vwf_r = fragalloc(2);
  void* nm1f_l = fragalloc(8); void* nm1f_r = fragalloc(8);
  void* nm2f_l = fragalloc(2); void* nm2f_r = fragalloc(2);

  zero_kernel<<<(int)((zero_floats + 255) / 256), 256, 0, stream>>>((float*)d_ws, (int)zero_floats);

  f2bf_kernel<<<(N_LIG * 64 + 255) / 256, 256, 0, stream>>>(h_lig, hbf_l, N_LIG * 64);
  f2bf_kernel<<<(N_REC * 64 + 255) / 256, 256, 0, stream>>>(h_rec, hbf_r, N_REC * 64);

  auto packW = [&](const void* W, int Korig, int ktiles, void* dst) {
    int total = ktiles * 2048;
    pack_wfrag<<<(total + 255) / 256, 256, 0, stream>>>((const float*)W, Korig, ktiles,
                                                        (__bf16*)dst);
  };
  packW(d_in[15], 158, 6, w1f_l);
  packW(d_in[21], 170, 6, w1f_r);
  packW(d_in[19], 64, 2, w2f_l);
  packW(d_in[25], 64, 2, w2f_r);
  packW(d_in[27], 64, 2, cw1f_l);
  packW(d_in[31], 64, 2, cw1f_r);
  packW(d_in[35], 64, 2, qwf_l);
  packW(d_in[36], 64, 2, kwf_l);
  packW(d_in[37], 64, 2, vwf_l);
  packW(d_in[38], 64, 2, qwf_r);
  packW(d_in[39], 64, 2, kwf_r);
  packW(d_in[40], 64, 2, vwf_r);
  packW(d_in[41], 256, 8, nm1f_l);
  packW(d_in[47], 256, 8, nm1f_r);
  packW(d_in[45], 64, 2, nm2f_l);
  packW(d_in[51], 64, 2, nm2f_r);

  qkv_kernel<<<N_LIG / 16, 32, 0, stream>>>(hbf_l, N_LIG, (const v16bf*)qwf_l,
      (const v16bf*)kwf_l, (const v16bf*)vwf_l, q_l, k_l, vT_l);
  qkv_kernel<<<N_REC / 16, 32, 0, stream>>>(hbf_r, N_REC, (const v16bf*)qwf_r,
      (const v16bf*)kwf_r, (const v16bf*)vwf_r, q_r, k_r, vT_r);

  attn_kernel<<<N_LIG / 16, 32, 0, stream>>>(q_l, k_r, vT_r, N_REC, mask,
      (long long)N_REC, 1LL, att_l);
  attn_kernel<<<N_REC / 16, 32, 0, stream>>>(q_r, k_l, vT_l, N_LIG, mask,
      1LL, (long long)N_REC, att_r);

  edge_kernel<<<E_LIG / 16, 32, 0, stream>>>(F_LIG, lig_src, lig_dst, lig_efeat, hbf_l,
      coords_lig, (const v16bf*)w1f_l, 6, (const float*)d_in[16], (const float*)d_in[17],
      (const float*)d_in[18], (const v16bf*)w2f_l, (const float*)d_in[20],
      (const v16bf*)cw1f_l, (const float*)d_in[28], (const float*)d_in[29],
      (const float*)d_in[30], aggr_l, xacc_l, cnt_l);
  edge_kernel<<<E_REC / 16, 32, 0, stream>>>(F_REC, rec_src, rec_dst, rec_efeat, hbf_r,
      coords_rec, (const v16bf*)w1f_r, 6, (const float*)d_in[22], (const float*)d_in[23],
      (const float*)d_in[24], (const v16bf*)w2f_r, (const float*)d_in[26],
      (const v16bf*)cw1f_r, (const float*)d_in[32], (const float*)d_in[33],
      (const float*)d_in[34], aggr_r, xacc_r, cnt_r);

  float* out = (float*)d_out;
  float* x_lig_out = out;                 // 4096*3
  float* h_lig_out = out + 12288;         // 4096*64
  float* x_rec_out = out + 274432;        // 8192*3
  float* h_rec_out = out + 299008;        // 8192*64

  coords_kernel<<<(N_LIG * 3 + 255) / 256, 256, 0, stream>>>(N_LIG, orig_c_lig, coords_lig,
      xacc_l, cnt_l, x_lig_out);
  coords_kernel<<<(N_REC * 3 + 255) / 256, 256, 0, stream>>>(N_REC, orig_c_rec, coords_rec,
      xacc_r, cnt_r, x_rec_out);

  node_kernel<<<N_LIG / 16, 32, 0, stream>>>(h_lig, aggr_l, cnt_l, att_l, orig_h_lig,
      (const v16bf*)nm1f_l, (const float*)d_in[42], (const float*)d_in[43],
      (const float*)d_in[44], (const v16bf*)nm2f_l, (const float*)d_in[46], h_lig_out);
  node_kernel<<<N_REC / 16, 32, 0, stream>>>(h_rec, aggr_r, cnt_r, att_r, orig_h_rec,
      (const v16bf*)nm1f_r, (const float*)d_in[48], (const float*)d_in[49],
      (const float*)d_in[50], (const v16bf*)nm2f_r, (const float*)d_in[52], h_rec_out);
}
